// CWRNN_75376676044995
// MI455X (gfx1250) — matrix operand order repaired and checked by reading, not compiled, not run on hardware
//
#include <hip/hip_runtime.h>
#include <stddef.h>

// ---------------------------------------------------------------------------
// CWRNN on gfx1250: persistent batch-tile workgroups, bf16 WMMA recurrence.
//   B=1024, T=256, DX=16, H=1024, DH2=512, DY=1
//   64 WGs x 16 batch rows; 8 waves/WG, wave w owns H-columns [128w,128w+128)
//   K augmented to 1056 = 33 k-tiles of 32 (tile 32 = masked x_t, zero-padded)
// ---------------------------------------------------------------------------

typedef __bf16 v16bf __attribute__((ext_vector_type(16)));
typedef float  v8f   __attribute__((ext_vector_type(8)));

#define T_STEPS 256
#define DXC     16
#define HC      1024
#define KT_ALL  33          // 32 k-tiles of h + 1 k-tile of x (padded)
#define NT_H    64          // 1024 / 16 output tiles of recurrence
#define DH2C    512
#define NT_Y    32          // 512 / 16 output tiles of MLP-1
#define KT_W1   32
#define HSTR    1064        // bf16 row stride in LDS: 2128B = 16*133 (odd*16 -> no bank camp)
#define A1STR   520         // f32 row stride for relu activations (reuses same LDS)

union Frag { v16bf v; uint4 q[2]; unsigned short s[16]; };

__device__ __forceinline__ unsigned short f2bf(float f) {
    union { float f; unsigned int u; } c; c.f = f;
    unsigned int u = c.u;
    u += 0x7FFFu + ((u >> 16) & 1u);          // round-to-nearest-even
    return (unsigned short)(u >> 16);
}

__device__ __forceinline__ float fast_tanh(float v) {
    float e = __expf(-2.0f * fabsf(v));       // v_exp_f32
    float r = __fdividef(1.0f - e, 1.0f + e);
    return copysignf(r, v);
}

// ---------------------------------------------------------------------------
// Pack W_aug = [W_h ; W_x ; 0] (1056 x 1024) and W1 (1024 x 512) into bf16 in
// this file's WMMA B-operand lane layout (contiguous K, same as A): per (kt,nt) tile of 32x16, lane L<16
// holds K = kt*32+0..15 of column nt*16+L, lanes 16..31 hold K = kt*32+16..31.
// Each lane owns 16 bf16 = 32 contiguous bytes -> runtime B load is 2x b128.
// ---------------------------------------------------------------------------
__global__ void cwrnn_pack(const float* __restrict__ w_x,
                           const float* __restrict__ w_h,
                           const float* __restrict__ w1,
                           unsigned short* __restrict__ wpk,
                           unsigned short* __restrict__ w1pk) {
    const int b    = blockIdx.x;
    const int lane = threadIdx.x;             // 0..31
    Frag o;
    if (b < KT_ALL * NT_H) {
        const int kt = b / NT_H, nt = b % NT_H;
        const int n  = nt * 16 + (lane & 15);
        const int k0 = kt * 32 + ((lane >> 4) << 4);
        #pragma unroll
        for (int j = 0; j < 16; ++j) {
            const int k = k0 + j;
            float v = 0.0f;
            if (k < HC)            v = w_h[(size_t)k * HC + n];
            else if (k < HC + DXC) v = w_x[(size_t)(k - HC) * HC + n];
            o.s[j] = f2bf(v);
        }
        uint4* dst = (uint4*)(wpk + (size_t)b * 512 + lane * 16);
        dst[0] = o.q[0]; dst[1] = o.q[1];
    } else {
        const int bb = b - KT_ALL * NT_H;
        const int kt = bb / NT_Y, nt = bb % NT_Y;
        const int n  = nt * 16 + (lane & 15);
        const int k0 = kt * 32 + ((lane >> 4) << 4);
        #pragma unroll
        for (int j = 0; j < 16; ++j)
            o.s[j] = f2bf(w1[(size_t)(k0 + j) * DH2C + n]);
        uint4* dst = (uint4*)(w1pk + (size_t)bb * 512 + lane * 16);
        dst[0] = o.q[0]; dst[1] = o.q[1];
    }
}

// ---------------------------------------------------------------------------
// Persistent recurrence + MLP head. h kept in LDS as bf16 row-major:
// A-operand per k-tile = 2x ds_load_b128 per lane (contiguous-K A layout used here:
// lanes 0-15 -> row m, K 0..15; lanes 16-31 -> row m, K 16..31).
// ---------------------------------------------------------------------------
__global__ __launch_bounds__(256) void cwrnn_main(
    const float* __restrict__ x,
    const unsigned short* __restrict__ wpk,
    const unsigned short* __restrict__ w1pk,
    const float* __restrict__ b1,
    const float* __restrict__ w2,
    const float* __restrict__ b2,
    float* __restrict__ y) {

    __shared__ __attribute__((aligned(16))) unsigned char smem[16 * HSTR * 2]; // 34,048 B
    unsigned short* sh  = (unsigned short*)smem;  // bf16 h state [16][HSTR]
    float*          sa1 = (float*)smem;           // reused for relu acts [16][A1STR]

    const int tid  = threadIdx.x;
    const int lane = tid & 31;
    const int wv   = tid >> 5;                    // 0..7
    const int m    = lane & 15;
    const int half = lane >> 4;
    const int wg   = blockIdx.x;                  // batch rows [wg*16, wg*16+16)

    // h0 = 0 (also zero-pads x k-tile columns 1040..1055)
    for (int i = tid; i < 16 * HSTR; i += 256) sh[i] = 0;
    __syncthreads();

    const uint4* wq = (const uint4*)wpk;
    v8f acc[8];

    for (int t = 0; t < T_STEPS; ++t) {
        // ---- stage masked x_t into augmented-K columns [1024,1040) ----
        {
            const int row = tid >> 4, col = tid & 15;
            const int step = t + 1;
            const int lim  = ((step & 7) == 0) ? 16 : (((step & 3) == 0) ? 12 : 8);
            float v = 0.0f;
            if (col < lim)
                v = x[((size_t)(wg * 16 + row) * T_STEPS + t) * DXC + col];
            sh[row * HSTR + HC + col] = f2bf(v);
        }
        __syncthreads();

        // clockwork: wave-uniform activity (h2 cols on step%4, h3 cols on step%8)
        const int u2 = (((t + 1) & 3) == 0);
        const int u3 = (((t + 1) & 7) == 0);
        const int act = (wv < 4) ? 1 : ((wv < 6) ? u2 : u3);

        if (__builtin_amdgcn_readfirstlane(act)) {
            #pragma unroll
            for (int j = 0; j < 8; ++j) {
                v8f z = {0.f,0.f,0.f,0.f,0.f,0.f,0.f,0.f};
                acc[j] = z;
            }
            for (int kt = 0; kt < KT_ALL; ++kt) {
                Frag a;
                const uint4* ap =
                    (const uint4*)(sh + m * HSTR + kt * 32 + half * 16);
                a.q[0] = ap[0];        // K +0..7   (of this lane's half)
                a.q[1] = ap[1];        // K +8..15
                #pragma unroll
                for (int j = 0; j < 8; ++j) {
                    const int nt = wv * 8 + j;
                    Frag bfr;
                    const uint4* bp = wq + (size_t)(kt * NT_H + nt) * 64 + lane * 2;
                    bfr.q[0] = bp[0]; bfr.q[1] = bp[1];
                    acc[j] = __builtin_amdgcn_wmma_f32_16x16x32_bf16(
                        false, a.v, false, bfr.v, (short)0, acc[j], false, false);
                }
            }
            #pragma unroll
            for (int j = 0; j < 8; ++j)
                #pragma unroll
                for (int v = 0; v < 8; ++v)
                    acc[j][v] = fast_tanh(acc[j][v]);
        }
        __syncthreads();   // all A reads done before h overwrite

        if (__builtin_amdgcn_readfirstlane(act)) {
            // D layout -> row-major bf16 h: lane = col, VGPR v = row v + 8*half
            #pragma unroll
            for (int j = 0; j < 8; ++j) {
                const int n = (wv * 8 + j) * 16 + m;
                #pragma unroll
                for (int v = 0; v < 8; ++v)
                    sh[(v + half * 8) * HSTR + n] = f2bf(acc[j][v]);
            }
        }
        // next iter's first barrier orders these writes vs. next A reads
    }
    __syncthreads();

    // ---- MLP stage 1: a1 = relu(h @ W1 + b1), 4 n-tiles per wave ----
    v8f acc2[4];
    #pragma unroll
    for (int j = 0; j < 4; ++j) {
        v8f z = {0.f,0.f,0.f,0.f,0.f,0.f,0.f,0.f};
        acc2[j] = z;
    }
    const uint4* w1q = (const uint4*)w1pk;
    for (int kt = 0; kt < KT_W1; ++kt) {
        Frag a;
        const uint4* ap = (const uint4*)(sh + m * HSTR + kt * 32 + half * 16);
        a.q[0] = ap[0];
        a.q[1] = ap[1];
        #pragma unroll
        for (int j = 0; j < 4; ++j) {
            const int nt = wv * 4 + j;
            Frag bfr;
            const uint4* bp = w1q + (size_t)(kt * NT_Y + nt) * 64 + lane * 2;
            bfr.q[0] = bp[0]; bfr.q[1] = bp[1];
            acc2[j] = __builtin_amdgcn_wmma_f32_16x16x32_bf16(
                false, a.v, false, bfr.v, (short)0, acc2[j], false, false);
        }
    }
    __syncthreads();   // h reads finished; safe to reuse LDS for a1

    #pragma unroll
    for (int j = 0; j < 4; ++j) {
        const int n = (wv * 4 + j) * 16 + m;
        const float bias = b1[n];
        #pragma unroll
        for (int v = 0; v < 8; ++v) {
            float r = acc2[j][v] + bias;
            sa1[(v + half * 8) * A1STR + n] = fmaxf(r, 0.0f);
        }
    }
    __syncthreads();

    // ---- MLP stage 2: y = a1 @ w2 + b2 (DY = 1) ----
    if (tid < 16) {
        float s = b2[0];
        for (int j = 0; j < DH2C; ++j)
            s = fmaf(sa1[tid * A1STR + j], w2[j], s);
        y[wg * 16 + tid] = s;
    }
}

// ---------------------------------------------------------------------------
extern "C" void kernel_launch(void* const* d_in, const int* in_sizes, int n_in,
                              void* d_out, int out_size, void* d_ws, size_t ws_size,
                              hipStream_t stream) {
    (void)in_sizes; (void)n_in; (void)out_size; (void)ws_size;
    const float* x   = (const float*)d_in[0];
    const float* w_x = (const float*)d_in[1];
    const float* w_h = (const float*)d_in[2];
    const float* w1  = (const float*)d_in[3];
    const float* b1  = (const float*)d_in[4];
    const float* w2  = (const float*)d_in[5];
    const float* b2  = (const float*)d_in[6];
    float* y = (float*)d_out;

    // workspace: packed bf16 weights (fully rewritten every call -> deterministic)
    unsigned short* wpk  = (unsigned short*)d_ws;                 // 33*64*512 bf16 = 2.06 MB
    unsigned short* w1pk = wpk + (size_t)KT_ALL * NT_H * 512;     // 32*32*512 bf16 = 1.0 MB

    cwrnn_pack<<<KT_ALL * NT_H + KT_W1 * NT_Y, 32, 0, stream>>>(w_x, w_h, w1, wpk, w1pk);
    cwrnn_main<<<64, 256, 0, stream>>>(x, wpk, w1pk, b1, w2, b2, y);
}
